// MultiheadAttention_65171833750065
// MI455X (gfx1250) — matrix operand order repaired and checked by reading, not compiled, not run on hardware
//
#include <hip/hip_runtime.h>

#define NCTX  4096
#define WIDTH 1024
#define HEADS 16
#define HDIM  64
#define QKVN  3072
#define QK_SCALE 0.35355339059327379f   // 1 / 64^(1/4)

typedef float  v8f     __attribute__((ext_vector_type(8)));
typedef __bf16 bf16x16 __attribute__((ext_vector_type(16)));
typedef __bf16 bf16x8  __attribute__((ext_vector_type(8)));
typedef __bf16 bf16x4  __attribute__((ext_vector_type(4)));
typedef __bf16 bf16x2  __attribute__((ext_vector_type(2)));

union FragU { bf16x16 v; bf16x8 h[2]; };

// A-matrix fragment (16x32 bf16, MxK): lane<16 holds row=lane, K {koff..+7, koff+16..+23}
__device__ __forceinline__ bf16x16 frag_a(const __bf16* base, int ld) {
  const int lane = threadIdx.x & 31;
  const int row  = lane & 15;
  const int koff = (lane >> 4) << 3;          // 0 or 8
  const __bf16* p = base + row * ld + koff;
  FragU f;
  f.h[0] = *(const bf16x8*)(p);
  f.h[1] = *(const bf16x8*)(p + 16);
  return f.v;
}

// B-matrix fragment (32x16 bf16, KxN), LDS stored [n][k]: lane<16 holds col=lane, K {koff..+7, koff+16..+23}
__device__ __forceinline__ bf16x16 frag_b(const __bf16* base, int ld) {
  const int lane = threadIdx.x & 31;
  const int col  = lane & 15;
  const int koff = (lane >> 4) << 3;          // 0 or 8
  const __bf16* p = base + col * ld + koff;
  FragU f;
  f.h[0] = *(const bf16x8*)(p);
  f.h[1] = *(const bf16x8*)(p + 16);
  return f.v;
}

__device__ __forceinline__ v8f wmma_bf16(bf16x16 a, bf16x16 b, v8f c) {
  return __builtin_amdgcn_wmma_f32_16x16x32_bf16(false, a, false, b, (short)0, c, false, false);
}

__device__ __forceinline__ v8f v8f_zero() {
  v8f z;
#pragma unroll
  for (int e = 0; e < 8; ++e) z[e] = 0.0f;
  return z;
}

// ---------------------------------------------------------------------------
// Kernel 1: qkv = x @ Wqkv + bqkv ; scale q,k columns ; bf16 out
// 128x256 block tile, 64x64 wave tile, BK=32, double-buffered LDS, reg prefetch
// ---------------------------------------------------------------------------
__global__ __launch_bounds__(256) void qkv_gemm_kernel(
    const float* __restrict__ X, const float* __restrict__ W,
    const float* __restrict__ bias, __bf16* __restrict__ out) {
  constexpr int BM = 128, BN = 256, BK = 32, LD = 40;
  __shared__ __align__(16) __bf16 As[2][BM * LD];   // 2 x 10240 B
  __shared__ __align__(16) __bf16 Bs[2][BN * LD];   // 2 x 20480 B

  const int tid = threadIdx.x;
  const int m0 = blockIdx.y * BM;
  const int n0 = blockIdx.x * BN;
  const int wid = tid >> 5;
  const int wm = (wid & 1) * 64;
  const int wn = (wid >> 1) * 64;

  const int aj  = tid & 7;          // A k-chunk (x4 floats)
  const int ai  = tid >> 3;         // A row 0..31 (+32r)
  const int bn4 = (tid & 63) * 4;   // B n
  const int bkb = (tid >> 6) * 2;   // B k pair base (+8r)

  v8f acc[4][4];
#pragma unroll
  for (int mi = 0; mi < 4; ++mi)
#pragma unroll
    for (int ni = 0; ni < 4; ++ni) acc[mi][ni] = v8f_zero();

  float4 apf[4];
  float4 bpf[4][2];

  auto load_pf = [&](int k0) {
#pragma unroll
    for (int r = 0; r < 4; ++r)
      apf[r] = *(const float4*)(X + (size_t)(m0 + ai + 32 * r) * WIDTH + k0 + aj * 4);
#pragma unroll
    for (int r = 0; r < 4; ++r) {
      const int k = bkb + 8 * r;
      bpf[r][0] = *(const float4*)(W + (size_t)(k0 + k)     * QKVN + n0 + bn4);
      bpf[r][1] = *(const float4*)(W + (size_t)(k0 + k + 1) * QKVN + n0 + bn4);
    }
  };
  auto store_pf = [&](int buf) {
#pragma unroll
    for (int r = 0; r < 4; ++r) {
      bf16x4 b;
      b[0] = (__bf16)apf[r].x; b[1] = (__bf16)apf[r].y;
      b[2] = (__bf16)apf[r].z; b[3] = (__bf16)apf[r].w;
      *(bf16x4*)&As[buf][(ai + 32 * r) * LD + aj * 4] = b;
    }
#pragma unroll
    for (int r = 0; r < 4; ++r) {
      const int k = bkb + 8 * r;
      bf16x2 p;
      p[0] = (__bf16)bpf[r][0].x; p[1] = (__bf16)bpf[r][1].x; *(bf16x2*)&Bs[buf][(bn4 + 0) * LD + k] = p;
      p[0] = (__bf16)bpf[r][0].y; p[1] = (__bf16)bpf[r][1].y; *(bf16x2*)&Bs[buf][(bn4 + 1) * LD + k] = p;
      p[0] = (__bf16)bpf[r][0].z; p[1] = (__bf16)bpf[r][1].z; *(bf16x2*)&Bs[buf][(bn4 + 2) * LD + k] = p;
      p[0] = (__bf16)bpf[r][0].w; p[1] = (__bf16)bpf[r][1].w; *(bf16x2*)&Bs[buf][(bn4 + 3) * LD + k] = p;
    }
  };

  constexpr int NK = WIDTH / BK;   // 32
  load_pf(0);
  store_pf(0);
  load_pf(BK);
  __syncthreads();

  for (int kt = 0; kt < NK; ++kt) {
    const int buf = kt & 1;
    bf16x16 a[4], b[4];
#pragma unroll
    for (int mi = 0; mi < 4; ++mi) a[mi] = frag_a(&As[buf][(wm + mi * 16) * LD], LD);
#pragma unroll
    for (int ni = 0; ni < 4; ++ni) b[ni] = frag_b(&Bs[buf][(wn + ni * 16) * LD], LD);
#pragma unroll
    for (int mi = 0; mi < 4; ++mi)
#pragma unroll
      for (int ni = 0; ni < 4; ++ni) acc[mi][ni] = wmma_bf16(a[mi], b[ni], acc[mi][ni]);
    if (kt + 1 < NK) {
      store_pf((kt + 1) & 1);                 // fill the other buffer
      if (kt + 2 < NK) load_pf((kt + 2) * BK);
    }
    __syncthreads();
  }

  const int lane = tid & 31;
  const int rsel = (lane >> 4) * 8;
  const int csel = lane & 15;
#pragma unroll
  for (int ni = 0; ni < 4; ++ni) {
    const int col = n0 + wn + ni * 16 + csel;
    const float bb = bias[col];
    const bool sc = (col % 192) < 128;        // q,k channels get the scale
#pragma unroll
    for (int mi = 0; mi < 4; ++mi)
#pragma unroll
      for (int r = 0; r < 8; ++r) {
        const int row = m0 + wm + mi * 16 + r + rsel;
        float v = acc[mi][ni][r] + bb;
        if (sc) v *= QK_SCALE;
        out[(size_t)row * QKVN + col] = (__bf16)v;
      }
  }
}

// ---------------------------------------------------------------------------
// Kernel 2: flash attention per head (bf16 in, bf16 out)
// double-buffered K/V LDS (Q staging overlapped onto buffer 1), reg prefetch
// ---------------------------------------------------------------------------
__global__ __launch_bounds__(256) void attn_fwd_kernel(
    const __bf16* __restrict__ qkv, __bf16* __restrict__ out) {
  constexpr int BQ = 128, BKV = 64, LK = 72, LV = 72, LP = 72;
  constexpr int KVELT = BKV * LK;                       // 4608 elts per K or V buffer
  __shared__ __align__(16) __bf16 smem[4 * KVELT + 8 * 16 * LP];  // 55296 B
  __bf16* const Ksb0 = smem;
  __bf16* const Vsb0 = smem + KVELT;
  __bf16* const Ksb1 = smem + 2 * KVELT;
  __bf16* const Vsb1 = smem + 3 * KVELT;
  __bf16* const Qs   = smem + 2 * KVELT;                // overlaps buf1: prologue-only
  __bf16* const Ps   = smem + 4 * KVELT;

  const int tid  = threadIdx.x;
  const int h    = blockIdx.y;
  const int q0   = blockIdx.x * BQ;
  const int wid  = tid >> 5;
  const int lane = tid & 31;
  const int rsel = (lane >> 4) * 8;
  const int csel = lane & 15;
  const int chunk = tid & 7;    // x8 channels
  const int lrow  = tid >> 3;   // 0..31

  bf16x8 kc[2], vc[2];
  auto load_kv = [&](int s0) {
#pragma unroll
    for (int r = 0; r < 2; ++r)
      kc[r] = *(const bf16x8*)(qkv + (size_t)(s0 + lrow + 32 * r) * QKVN + h * 192 + 64 + chunk * 8);
#pragma unroll
    for (int r = 0; r < 2; ++r)
      vc[r] = *(const bf16x8*)(qkv + (size_t)(s0 + 2 * lrow + r) * QKVN + h * 192 + 128 + chunk * 8);
  };
  auto store_kv = [&](__bf16* Ks, __bf16* Vs) {
#pragma unroll
    for (int r = 0; r < 2; ++r)
      *(bf16x8*)&Ks[(lrow + 32 * r) * LK + chunk * 8] = kc[r];
#pragma unroll
    for (int e = 0; e < 8; ++e) {             // V transposed [ch][key], key pairs packed
      bf16x2 p; p[0] = vc[0][e]; p[1] = vc[1][e];
      *(bf16x2*)&Vs[(chunk * 8 + e) * LV + 2 * lrow] = p;
    }
  };

  // ---- prologue: Q tile + KV tile 0 ----
#pragma unroll
  for (int r = 0; r < 4; ++r) {
    const int row = lrow + 32 * r;
    *(bf16x8*)&Qs[row * LK + chunk * 8] =
        *(const bf16x8*)(qkv + (size_t)(q0 + row) * QKVN + h * 192 + chunk * 8);
  }
  load_kv(0);
  __syncthreads();
  const bf16x16 qf0 = frag_a(&Qs[(wid * 16) * LK + 0],  LK);
  const bf16x16 qf1 = frag_a(&Qs[(wid * 16) * LK + 32], LK);
  store_kv(Ksb0, Vsb0);                       // buf0 (disjoint from Qs region)
  load_kv(BKV);
  __syncthreads();

  float mstate[8], lstate[8];
  v8f o[4];
#pragma unroll
  for (int r = 0; r < 8; ++r) { mstate[r] = -1e30f; lstate[r] = 0.0f; }
#pragma unroll
  for (int n = 0; n < 4; ++n) o[n] = v8f_zero();

  __bf16* const Pw = &Ps[wid * 16 * LP];
  constexpr int NT = NCTX / BKV;   // 64

  for (int t = 0; t < NT; ++t) {
    const __bf16* Ks = (t & 1) ? Ksb1 : Ksb0;
    const __bf16* Vs = (t & 1) ? Vsb1 : Vsb0;

    // ---- S = Q K^T : 16 rows x 64 keys per wave ----
    v8f s[4];
#pragma unroll
    for (int j = 0; j < 4; ++j) {
      s[j] = v8f_zero();
      s[j] = wmma_bf16(qf0, frag_b(&Ks[(j * 16) * LK + 0],  LK), s[j]);
      s[j] = wmma_bf16(qf1, frag_b(&Ks[(j * 16) * LK + 32], LK), s[j]);
    }

    // ---- online softmax (rows r / r+8 per lane-half) ----
#pragma unroll
    for (int r = 0; r < 8; ++r) {
      float mx = s[0][r];
#pragma unroll
      for (int j = 1; j < 4; ++j) mx = fmaxf(mx, s[j][r]);
#pragma unroll
      for (int off = 8; off >= 1; off >>= 1) mx = fmaxf(mx, __shfl_xor(mx, off, 32));
      const float mnew = fmaxf(mstate[r], mx);
      const float cf   = __expf(mstate[r] - mnew);
      mstate[r] = mnew;
      float ps = 0.0f;
#pragma unroll
      for (int j = 0; j < 4; ++j) {
        const float p = __expf(s[j][r] - mnew);
        ps += p;
        Pw[(r + rsel) * LP + j * 16 + csel] = (__bf16)p;
      }
#pragma unroll
      for (int off = 8; off >= 1; off >>= 1) ps += __shfl_xor(ps, off, 32);
      lstate[r] = lstate[r] * cf + ps;
#pragma unroll
      for (int n = 0; n < 4; ++n) o[n][r] *= cf;
    }

    // ---- O += P @ V ----
    bf16x16 pf[2];
#pragma unroll
    for (int kk = 0; kk < 2; ++kk) pf[kk] = frag_a(&Pw[kk * 32], LP);
#pragma unroll
    for (int n = 0; n < 4; ++n)
#pragma unroll
      for (int kk = 0; kk < 2; ++kk)
        o[n] = wmma_bf16(pf[kk], frag_b(&Vs[(n * 16) * LV + kk * 32], LV), o[n]);

    // ---- commit prefetched tile into the other buffer, prefetch t+2 ----
    if (t + 1 < NT) {
      if (t & 1) store_kv(Ksb0, Vsb0); else store_kv(Ksb1, Vsb1);
      if (t + 2 < NT) load_kv((t + 2) * BKV);
    }
    __syncthreads();
  }

  // ---- epilogue: O / l -> bf16 ----
#pragma unroll
  for (int n = 0; n < 4; ++n)
#pragma unroll
    for (int r = 0; r < 8; ++r) {
      const int row = q0 + wid * 16 + r + rsel;
      const int c   = h * HDIM + n * 16 + csel;
      out[(size_t)row * WIDTH + c] = (__bf16)(o[n][r] / lstate[r]);
    }
}

// ---------------------------------------------------------------------------
// Kernel 3: out = attn @ Wproj + bproj (fp32 out)
// 128x128 block tile, BK=32, double-buffered, reg prefetch
// ---------------------------------------------------------------------------
__global__ __launch_bounds__(256) void proj_gemm_kernel(
    const __bf16* __restrict__ A, const float* __restrict__ W,
    const float* __restrict__ bias, float* __restrict__ out) {
  constexpr int BM = 128, BN = 128, BK = 32, LD = 40;
  __shared__ __align__(16) __bf16 As[2][BM * LD];
  __shared__ __align__(16) __bf16 Bs[2][BN * LD];

  const int tid = threadIdx.x;
  const int m0 = blockIdx.y * BM;
  const int n0 = blockIdx.x * BN;
  const int wid = tid >> 5;
  const int wm = (wid & 1) * 64;
  const int wn = (wid >> 1) * 32;

  const int ac  = tid & 3;          // A k-chunk (x8 bf16)
  const int ar  = tid >> 2;         // A row 0..63 (+64r)
  const int bn4 = (tid & 31) * 4;   // B n
  const int bkb = (tid >> 5) * 2;   // B k pair base (+16r)

  v8f acc[4][2];
#pragma unroll
  for (int mi = 0; mi < 4; ++mi)
#pragma unroll
    for (int ni = 0; ni < 2; ++ni) acc[mi][ni] = v8f_zero();

  bf16x8 apf[2];
  float4 bpf[2][2];

  auto load_pf = [&](int k0) {
#pragma unroll
    for (int r = 0; r < 2; ++r)
      apf[r] = *(const bf16x8*)(A + (size_t)(m0 + ar + 64 * r) * WIDTH + k0 + ac * 8);
#pragma unroll
    for (int r = 0; r < 2; ++r) {
      const int k = bkb + 16 * r;
      bpf[r][0] = *(const float4*)(W + (size_t)(k0 + k)     * WIDTH + n0 + bn4);
      bpf[r][1] = *(const float4*)(W + (size_t)(k0 + k + 1) * WIDTH + n0 + bn4);
    }
  };
  auto store_pf = [&](int buf) {
#pragma unroll
    for (int r = 0; r < 2; ++r)
      *(bf16x8*)&As[buf][(ar + 64 * r) * LD + ac * 8] = apf[r];
#pragma unroll
    for (int r = 0; r < 2; ++r) {
      const int k = bkb + 16 * r;
      bf16x2 p;
      p[0] = (__bf16)bpf[r][0].x; p[1] = (__bf16)bpf[r][1].x; *(bf16x2*)&Bs[buf][(bn4 + 0) * LD + k] = p;
      p[0] = (__bf16)bpf[r][0].y; p[1] = (__bf16)bpf[r][1].y; *(bf16x2*)&Bs[buf][(bn4 + 1) * LD + k] = p;
      p[0] = (__bf16)bpf[r][0].z; p[1] = (__bf16)bpf[r][1].z; *(bf16x2*)&Bs[buf][(bn4 + 2) * LD + k] = p;
      p[0] = (__bf16)bpf[r][0].w; p[1] = (__bf16)bpf[r][1].w; *(bf16x2*)&Bs[buf][(bn4 + 3) * LD + k] = p;
    }
  };

  constexpr int NK = WIDTH / BK;   // 32
  load_pf(0);
  store_pf(0);
  load_pf(BK);
  __syncthreads();

  for (int kt = 0; kt < NK; ++kt) {
    const int buf = kt & 1;
    bf16x16 a[4], b[2];
#pragma unroll
    for (int mi = 0; mi < 4; ++mi) a[mi] = frag_a(&As[buf][(wm + mi * 16) * LD], LD);
#pragma unroll
    for (int ni = 0; ni < 2; ++ni) b[ni] = frag_b(&Bs[buf][(wn + ni * 16) * LD], LD);
#pragma unroll
    for (int mi = 0; mi < 4; ++mi)
#pragma unroll
      for (int ni = 0; ni < 2; ++ni) acc[mi][ni] = wmma_bf16(a[mi], b[ni], acc[mi][ni]);
    if (kt + 1 < NK) {
      store_pf((kt + 1) & 1);
      if (kt + 2 < NK) load_pf((kt + 2) * BK);
    }
    __syncthreads();
  }

  const int lane = tid & 31;
  const int rsel = (lane >> 4) * 8;
  const int csel = lane & 15;
#pragma unroll
  for (int ni = 0; ni < 2; ++ni) {
    const int col = n0 + wn + ni * 16 + csel;
    const float bb = bias[col];
#pragma unroll
    for (int mi = 0; mi < 4; ++mi)
#pragma unroll
      for (int r = 0; r < 8; ++r) {
        const int row = m0 + wm + mi * 16 + r + rsel;
        out[(size_t)row * WIDTH + col] = acc[mi][ni][r] + bb;
      }
  }
}

// ---------------------------------------------------------------------------
extern "C" void kernel_launch(void* const* d_in, const int* in_sizes, int n_in,
                              void* d_out, int out_size, void* d_ws, size_t ws_size,
                              hipStream_t stream) {
  (void)in_sizes; (void)n_in; (void)out_size; (void)ws_size;
  const float* x     = (const float*)d_in[0];
  const float* Wqkv  = (const float*)d_in[1];
  const float* bqkv  = (const float*)d_in[2];
  const float* Wproj = (const float*)d_in[3];
  const float* bproj = (const float*)d_in[4];
  float* outp = (float*)d_out;

  __bf16* qkv_ws  = (__bf16*)d_ws;                          // 4096 x 3072 bf16 (24 MB)
  __bf16* attn_ws = qkv_ws + (size_t)NCTX * QKVN;           // 4096 x 1024 bf16 (8 MB)

  qkv_gemm_kernel<<<dim3(QKVN / 256, NCTX / 128), 256, 0, stream>>>(x, Wqkv, bqkv, qkv_ws);
  attn_fwd_kernel<<<dim3(NCTX / 128, HEADS), 256, 0, stream>>>(qkv_ws, attn_ws);
  proj_gemm_kernel<<<dim3(WIDTH / 128, NCTX / 128), 256, 0, stream>>>(attn_ws, Wproj, bproj, outp);
}
